// Attention_84516366450854
// MI455X (gfx1250) — matrix operand order repaired and checked by reading, not compiled, not run on hardware
//
#include <hip/hip_runtime.h>

typedef _Float16 half_t;
typedef __attribute__((ext_vector_type(16))) _Float16 v16h;
typedef __attribute__((ext_vector_type(8)))  _Float16 v8h;
typedef __attribute__((ext_vector_type(8)))  float    v8f;

#define B_ 2
#define S_ 2048
#define E_ 768
#define H_ 12
#define D_ 64
#define MTOT (B_*S_)                     // 4096 rows
#define BHSD ((size_t)B_*H_*S_*D_)

// workspace layout (in half_t units)
#define OFF_XH  ((size_t)0)
#define OFF_W   (OFF_XH + (size_t)MTOT*E_)        // 4 transposed f16 weights
#define OFF_Q   (OFF_W + (size_t)4*E_*E_)         // [b,h,s,d]
#define OFF_K   (OFF_Q + BHSD)                    // [b,h,s,d]
#define OFF_V   (OFF_K + BHSD)                    // transposed: [b,h,d,s]
#define OFF_O   (OFF_V + BHSD)                    // [b,s,E] f16

__device__ inline v16h cat16(v8h lo, v8h hh) {
  return __builtin_shufflevector(lo, hh, 0,1,2,3,4,5,6,7,8,9,10,11,12,13,14,15);
}
#define WMMA_F16(A,Bf,C) __builtin_amdgcn_wmma_f32_16x16x32_f16(false,(A),false,(Bf),(short)0,(C),false,false)

// ---------------- conversion kernels ----------------
__global__ void cvt_f16(const float* __restrict__ src, half_t* __restrict__ dst, int n) {
  int i = blockIdx.x * blockDim.x + threadIdx.x;
  if (i < n) dst[i] = (half_t)src[i];
}

// dst[n*E + k] = src[k*E + n]
__global__ void cvt_tr(const float* __restrict__ src, half_t* __restrict__ dst) {
  int i = blockIdx.x * blockDim.x + threadIdx.x;
  if (i >= E_ * E_) return;
  int n = i / E_, k = i % E_;
  dst[i] = (half_t)src[(size_t)k * E_ + n];
}

// ---------------- fused QKV projection ----------------
// grid: (E/64, MTOT/128, 3); block 128 = 4 waves, each wave 32x64, software-pipelined
__global__ __launch_bounds__(128) void gemm_qkv(
    const half_t* __restrict__ xh, const half_t* __restrict__ wt,
    const float* __restrict__ bq, const float* __restrict__ bk, const float* __restrict__ bvv,
    half_t* __restrict__ Qd, half_t* __restrict__ Kd, half_t* __restrict__ Vt)
{
  const int z    = blockIdx.z;
  const int lane = threadIdx.x & 31, wave = threadIdx.x >> 5;
  const int hi   = lane >> 4, ln = lane & 15;
  const int m0   = blockIdx.y * 128 + wave * 32;
  const int n0   = blockIdx.x * 64;
  const half_t* Bt   = wt + (size_t)z * E_ * E_;
  const float*  bias = (z == 0) ? bq : (z == 1) ? bk : bvv;
  const half_t* arow0 = xh + (size_t)(m0 + ln) * E_;
  const half_t* arow1 = xh + (size_t)(m0 + 16 + ln) * E_;

#define LOADA(dst, rowp, kk) { const half_t* ap_ = (rowp) + (kk) + hi * 16; \
    dst = cat16(*(const v8h*)ap_, *(const v8h*)(ap_ + 8)); }
#define LOADB(dst, j, kk) { const half_t* bp_ = Bt + (size_t)(n0 + (j) * 16 + ln) * E_ + (kk) + hi * 16; \
    dst = *(const v16h*)bp_; }

  v8f acc[2][4] = {};
  v16h a0, a1, b0, b1, b2, b3;
  LOADA(a0, arow0, 0) LOADA(a1, arow1, 0)
  LOADB(b0, 0, 0) LOADB(b1, 1, 0) LOADB(b2, 2, 0) LOADB(b3, 3, 0)
  for (int k = 0; k < E_; k += 32) {
    const int kn = (k + 32 < E_) ? k + 32 : 0;     // last-iter prefetch is dead but in-bounds
    v16h na0, na1, nb0, nb1, nb2, nb3;
    LOADA(na0, arow0, kn) LOADA(na1, arow1, kn)
    LOADB(nb0, 0, kn) LOADB(nb1, 1, kn) LOADB(nb2, 2, kn) LOADB(nb3, 3, kn)
    acc[0][0] = WMMA_F16(a0, b0, acc[0][0]);  acc[1][0] = WMMA_F16(a1, b0, acc[1][0]);
    acc[0][1] = WMMA_F16(a0, b1, acc[0][1]);  acc[1][1] = WMMA_F16(a1, b1, acc[1][1]);
    acc[0][2] = WMMA_F16(a0, b2, acc[0][2]);  acc[1][2] = WMMA_F16(a1, b2, acc[1][2]);
    acc[0][3] = WMMA_F16(a0, b3, acc[0][3]);  acc[1][3] = WMMA_F16(a1, b3, acc[1][3]);
    a0 = na0; a1 = na1; b0 = nb0; b1 = nb1; b2 = nb2; b3 = nb3;
  }
#pragma unroll
  for (int mt = 0; mt < 2; ++mt) {
#pragma unroll
    for (int j = 0; j < 4; ++j) {
      const int n = n0 + j * 16 + ln;
      const float bval = bias[n];
      const int hh = n >> 6, d = n & 63;
#pragma unroll
      for (int r = 0; r < 8; ++r) {
        const int m  = m0 + mt * 16 + r + 8 * hi;
        const int bb = m >> 11, sp = m & (S_ - 1);
        const float v = acc[mt][j][r] + bval;
        if (z == 0)      Qd[(((size_t)bb * H_ + hh) * S_ + sp) * D_ + d] = (half_t)v;
        else if (z == 1) Kd[(((size_t)bb * H_ + hh) * S_ + sp) * D_ + d] = (half_t)v;
        else             Vt[(((size_t)bb * H_ + hh) * D_ + d) * S_ + sp] = (half_t)v;
      }
    }
  }
}

// ---------------- RoPE (in place on Q,K) ----------------
__global__ void rope_qk(half_t* __restrict__ Qd, half_t* __restrict__ Kd) {
  int i = blockIdx.x * blockDim.x + threadIdx.x;
  int d  = i & 31;
  int t  = i >> 5;
  int sp = t & (S_ - 1);
  int bh = t >> 11;
  if (bh >= B_ * H_) return;
  float theta = (float)sp * __powf(10000.0f, -(float)d * (1.0f / 32.0f));
  float sn, cs;
  __sincosf(theta, &sn, &cs);
  size_t base = ((size_t)bh * S_ + sp) * D_;
  float q1 = (float)Qd[base + d], q2 = (float)Qd[base + d + 32];
  Qd[base + d]      = (half_t)(q1 * cs - q2 * sn);
  Qd[base + d + 32] = (half_t)(q2 * cs + q1 * sn);
  float k1 = (float)Kd[base + d], k2 = (float)Kd[base + d + 32];
  Kd[base + d]      = (half_t)(k1 * cs - k2 * sn);
  Kd[base + d + 32] = (half_t)(k2 * cs + k1 * sn);
}

// ---------------- flash-style causal attention ----------------
// grid: (S/64, H, B); block 128. K/V tiles staged to LDS with async copies,
// double-buffered; per-wave online softmax on 16 query rows.
__global__ __launch_bounds__(128) void attn(
    const half_t* __restrict__ Qd, const half_t* __restrict__ Kd,
    const half_t* __restrict__ Vt, half_t* __restrict__ O)
{
  const int lane = threadIdx.x & 31, wave = threadIdx.x >> 5;
  const int hi   = lane >> 4, ln = lane & 15;
  const int h = blockIdx.y, b = blockIdx.z;
  const int mbase = blockIdx.x * 64 + wave * 16;
  const half_t* Qp = Qd + (size_t)(b * H_ + h) * S_ * D_;
  const half_t* Kp = Kd + (size_t)(b * H_ + h) * S_ * D_;
  const half_t* Vp = Vt + (size_t)(b * H_ + h) * D_ * S_;

  __shared__ __align__(64) struct {
    _Float16 k[2][32][64];   // double-buffered K tile  (8 KB)
    _Float16 v[2][64][32];   // double-buffered V tile  (8 KB)
    _Float16 p[4][16][32];   // per-wave C->A reshape   (4 KB)
  } sm;

  // async cooperative stage of one 32-key block into buffer `buf`
  auto stage = [&](int kb, int buf) {
    const int t = threadIdx.x;
#pragma unroll
    for (int u = 0; u < 2; ++u) {                 // K tile: 256 16B chunks
      const int q = t + u * 128;
      unsigned loff = (unsigned)(size_t)&sm.k[buf][q >> 3][(q & 7) * 8];
      const half_t* g = Kp + (size_t)(kb + (q >> 3)) * D_ + (q & 7) * 8;
      asm volatile("global_load_async_to_lds_b128 %0, %1, off"
                   :: "v"(loff), "v"(g) : "memory");
    }
#pragma unroll
    for (int u = 0; u < 2; ++u) {                 // V tile: 256 16B chunks
      const int q = t + u * 128;
      unsigned loff = (unsigned)(size_t)&sm.v[buf][q >> 2][(q & 3) * 8];
      const half_t* g = Vp + (size_t)(q >> 2) * S_ + kb + (q & 3) * 8;
      asm volatile("global_load_async_to_lds_b128 %0, %1, off"
                   :: "v"(loff), "v"(g) : "memory");
    }
  };

  v16h aq[2];
  {
    const half_t* qrow = Qp + (size_t)(mbase + ln) * D_;
#pragma unroll
    for (int c = 0; c < 2; ++c) {
      const half_t* ap = qrow + c * 32 + hi * 16;
      aq[c] = cat16(*(const v8h*)ap, *(const v8h*)(ap + 8));
    }
  }
  v8f acc[4] = {};
  float mrun[8], lrun[8];
#pragma unroll
  for (int r = 0; r < 8; ++r) { mrun[r] = -3.0e38f; lrun[r] = 0.0f; }

  const float SCALE = 0.125f;            // 1/sqrt(64)
  const int Send = blockIdx.x * 64 + 64; // block-uniform causal key bound (exclusive)

  stage(0, 0);
  int it = 0;
  for (int kb = 0; kb < Send; kb += 32, ++it) {
    const int cur = it & 1;
    if (kb + 32 < Send) {
      stage(kb + 32, cur ^ 1);
      // 4 async instructions per wave per batch; in-order completion =>
      // <=4 outstanding means the current buffer's batch has landed.
      asm volatile("s_wait_asynccnt 0x4" ::: "memory");
    } else {
      asm volatile("s_wait_asynccnt 0x0" ::: "memory");
    }
    __syncthreads();                      // all waves' tile parts visible

    if (mbase + 15 >= kb) {               // wave has unmasked rows in this block
      v8f s[2];
#pragma unroll
      for (int ns = 0; ns < 2; ++ns) {
        v8f sacc = {};
#pragma unroll
        for (int c = 0; c < 2; ++c) {
          v16h bk = *(const v16h*)&sm.k[cur][ns * 16 + ln][c * 32 + hi * 16];
          sacc = WMMA_F16(aq[c], bk, sacc);
        }
        s[ns] = sacc;
      }
#pragma unroll
      for (int r = 0; r < 8; ++r) {
        const int mrow = mbase + r + 8 * hi;
        float s0 = s[0][r] * SCALE, s1 = s[1][r] * SCALE;
        if (kb + ln > mrow)      s0 = -3.0e38f;
        if (kb + 16 + ln > mrow) s1 = -3.0e38f;
        float bm = fmaxf(s0, s1);
#pragma unroll
        for (int off = 1; off < 16; off <<= 1)
          bm = fmaxf(bm, __shfl_xor(bm, off, 16));
        const float nm   = fmaxf(mrun[r], bm);
        const float corr = __expf(mrun[r] - nm);
        mrun[r] = nm;
        const float p0 = __expf(s0 - nm);
        const float p1 = __expf(s1 - nm);
        float rs = p0 + p1;
#pragma unroll
        for (int off = 1; off < 16; off <<= 1)
          rs += __shfl_xor(rs, off, 16);
        lrun[r] = lrun[r] * corr + rs;
#pragma unroll
        for (int dt = 0; dt < 4; ++dt) acc[dt][r] *= corr;
        sm.p[wave][r + 8 * hi][ln]      = (_Float16)p0;
        sm.p[wave][r + 8 * hi][16 + ln] = (_Float16)p1;
      }
      asm volatile("s_wait_dscnt 0x0" ::: "memory");
      const _Float16* pp = &sm.p[wave][ln][hi * 16];
      v16h pa = cat16(*(const v8h*)pp, *(const v8h*)(pp + 8));
#pragma unroll
      for (int dt = 0; dt < 4; ++dt) {
        v16h bv = *(const v16h*)&sm.v[cur][dt * 16 + ln][hi * 16];
        acc[dt] = WMMA_F16(pa, bv, acc[dt]);
      }
    }
    __syncthreads();                      // protect buffer reuse next iteration
  }
#pragma unroll
  for (int dt = 0; dt < 4; ++dt) {
    const int d = dt * 16 + ln;
#pragma unroll
    for (int r = 0; r < 8; ++r) {
      const int m = mbase + r + 8 * hi;
      O[((size_t)b * S_ + m) * E_ + h * D_ + d] = (half_t)(acc[dt][r] / lrun[r]);
    }
  }
}

// ---------------- output projection ----------------
// grid: (E/64, MTOT/128); block 128, wave tile 32x64, pipelined
__global__ __launch_bounds__(128) void gemm_out(
    const half_t* __restrict__ Oh, const half_t* __restrict__ Wpt,
    const float* __restrict__ bp, float* __restrict__ y)
{
  const int lane = threadIdx.x & 31, wave = threadIdx.x >> 5;
  const int hi   = lane >> 4, ln = lane & 15;
  const int m0 = blockIdx.y * 128 + wave * 32;
  const int n0 = blockIdx.x * 64;
  const half_t* arow0 = Oh + (size_t)(m0 + ln) * E_;
  const half_t* arow1 = Oh + (size_t)(m0 + 16 + ln) * E_;
  const half_t* Bt = Wpt;

  v8f acc[2][4] = {};
  v16h a0, a1, b0, b1, b2, b3;
  LOADA(a0, arow0, 0) LOADA(a1, arow1, 0)
  LOADB(b0, 0, 0) LOADB(b1, 1, 0) LOADB(b2, 2, 0) LOADB(b3, 3, 0)
  for (int k = 0; k < E_; k += 32) {
    const int kn = (k + 32 < E_) ? k + 32 : 0;
    v16h na0, na1, nb0, nb1, nb2, nb3;
    LOADA(na0, arow0, kn) LOADA(na1, arow1, kn)
    LOADB(nb0, 0, kn) LOADB(nb1, 1, kn) LOADB(nb2, 2, kn) LOADB(nb3, 3, kn)
    acc[0][0] = WMMA_F16(a0, b0, acc[0][0]);  acc[1][0] = WMMA_F16(a1, b0, acc[1][0]);
    acc[0][1] = WMMA_F16(a0, b1, acc[0][1]);  acc[1][1] = WMMA_F16(a1, b1, acc[1][1]);
    acc[0][2] = WMMA_F16(a0, b2, acc[0][2]);  acc[1][2] = WMMA_F16(a1, b2, acc[1][2]);
    acc[0][3] = WMMA_F16(a0, b3, acc[0][3]);  acc[1][3] = WMMA_F16(a1, b3, acc[1][3]);
    a0 = na0; a1 = na1; b0 = nb0; b1 = nb1; b2 = nb2; b3 = nb3;
  }
#pragma unroll
  for (int mt = 0; mt < 2; ++mt) {
#pragma unroll
    for (int j = 0; j < 4; ++j) {
      const int n = n0 + j * 16 + ln;
      const float bb = bp[n];
#pragma unroll
      for (int r = 0; r < 8; ++r) {
        const int m = m0 + mt * 16 + r + 8 * hi;
        y[(size_t)m * E_ + n] = acc[mt][j][r] + bb;
      }
    }
  }
}

// ---------------- launcher ----------------
extern "C" void kernel_launch(void* const* d_in, const int* in_sizes, int n_in,
                              void* d_out, int out_size, void* d_ws, size_t ws_size,
                              hipStream_t stream) {
  const float* x  = (const float*)d_in[0];
  const float* Wq = (const float*)d_in[1];
  const float* bq = (const float*)d_in[2];
  const float* Wk = (const float*)d_in[3];
  const float* bk = (const float*)d_in[4];
  const float* Wv = (const float*)d_in[5];
  const float* bv = (const float*)d_in[6];
  const float* Wp = (const float*)d_in[7];
  const float* bp = (const float*)d_in[8];
  float* y = (float*)d_out;

  half_t* ws = (half_t*)d_ws;
  half_t* xh = ws + OFF_XH;
  half_t* wt = ws + OFF_W;
  half_t* Qd = ws + OFF_Q;
  half_t* Kd = ws + OFF_K;
  half_t* Vt = ws + OFF_V;
  half_t* Oh = ws + OFF_O;

  {
    int n = MTOT * E_;
    cvt_f16<<<(n + 255) / 256, 256, 0, stream>>>(x, xh, n);
  }
  {
    int n = E_ * E_;
    int g = (n + 255) / 256;
    cvt_tr<<<g, 256, 0, stream>>>(Wq, wt + 0 * (size_t)E_ * E_);
    cvt_tr<<<g, 256, 0, stream>>>(Wk, wt + 1 * (size_t)E_ * E_);
    cvt_tr<<<g, 256, 0, stream>>>(Wv, wt + 2 * (size_t)E_ * E_);
    cvt_tr<<<g, 256, 0, stream>>>(Wp, wt + 3 * (size_t)E_ * E_);
  }
  gemm_qkv<<<dim3(E_ / 64, MTOT / 128, 3), 128, 0, stream>>>(xh, wt, bq, bk, bv, Qd, Kd, Vt);
  {
    int n = B_ * H_ * S_ * 32;
    rope_qk<<<(n + 255) / 256, 256, 0, stream>>>(Qd, Kd);
  }
  attn<<<dim3(S_ / 64, H_, B_), 128, 0, stream>>>(Qd, Kd, Vt, Oh);
  gemm_out<<<dim3(E_ / 64, MTOT / 128, 1), 128, 0, stream>>>(Oh, wt + 3 * (size_t)E_ * E_, bp, y);
}
